// CNNEncoder_26027501814586
// MI455X (gfx1250) — matrix operand order repaired and checked by reading, not compiled, not run on hardware
//
#include <hip/hip_runtime.h>

typedef __attribute__((ext_vector_type(16))) _Float16 v16h;
typedef __attribute__((ext_vector_type(8)))  float    v8f;

#define NB   4
#define NNODE 40
#define NT   100
#define NDIM 4
#define NH   128
#define NEDGE 1560            // edges per batch (N*(N-1))
#define NS   6240             // NB*NEDGE rows
#define TP   48               // pooled time length after conv1+maxpool
#define T2   44               // conv2 output time length
#define M2   (NS * T2)        // 274560 implicit-GEMM rows
#define MT2  (M2 / 16)        // 17160 M-tiles
#define BN_EPS 1e-5f

__device__ __forceinline__ float eluf(float x) { return x > 0.f ? x : __expf(x) - 1.f; }

// Edge e (NRI row-major off-diagonal enumeration): receiver = e/39, sender skips diagonal.
__device__ __forceinline__ void edge_nodes(int e, int& snd, int& rcv) {
  int r = e / (NNODE - 1);
  int k = e % (NNODE - 1);
  rcv = r;
  snd = (k < r) ? k : k + 1;
}

// A fragment (16x32 f16) for V_WMMA_F32_16X16X32_F16.
// p points at A[row= lane&15][kchunk*32]. Lane<16 holds K {0..15}, lane>=16 holds {16..31}.
__device__ __forceinline__ v16h load_a16(const _Float16* p, int lane) {
  const _Float16* q = p + ((lane & 16) ? 16 : 0);
  v16h a;
#pragma unroll
  for (int i = 0; i < 16; ++i) { a[i] = q[i]; }
  return a;
}

__device__ __forceinline__ v8f wmma16(v16h a, v16h b, v8f c) {
  return __builtin_amdgcn_wmma_f32_16x16x32_f16(false, a, false, b, (short)0, c, false, false);
}

// ---------------- weight swizzle kernels (B fragments, fragment-major fp16) ----------------
// dst[((c*8+nt)*32+lane)*16 + j] = B[c*32 + (lane>=16?16:0) + j][nt*16 + (lane&15)]

__global__ void k_swz_rm(const float* __restrict__ src, _Float16* __restrict__ dst, int K) {
  int idx = blockIdx.x * 256 + threadIdx.x;
  if (idx >= K * 128) return;
  int j = idx & 15, lane = (idx >> 4) & 31, nt = (idx >> 9) & 7, c = idx >> 12;
  int kg  = c * 32 + ((lane & 16) ? 16 : 0) + j;
  int col = nt * 16 + (lane & 15);
  dst[idx] = (_Float16)src[kg * 128 + col];          // src row-major [K][128]
}

__global__ void k_swz_tr(const float* __restrict__ src, _Float16* __restrict__ dst, int K) {
  int idx = blockIdx.x * 256 + threadIdx.x;
  if (idx >= K * 128) return;
  int j = idx & 15, lane = (idx >> 4) & 31, nt = (idx >> 9) & 7, c = idx >> 12;
  int kg  = c * 32 + ((lane & 16) ? 16 : 0) + j;
  int col = nt * 16 + (lane & 15);
  dst[idx] = (_Float16)src[col * K + kg];            // src [N=128][K] (e.g. convp_w[co][ci])
}

__global__ void k_swz_conv2(const float* __restrict__ w2, _Float16* __restrict__ dst) {
  int idx = blockIdx.x * 256 + threadIdx.x;
  if (idx >= 640 * 128) return;
  int j = idx & 15, lane = (idx >> 4) & 31, nt = (idx >> 9) & 7, c = idx >> 12;
  int kg  = c * 32 + ((lane & 16) ? 16 : 0) + j;     // kg = k*128 + ci
  int col = nt * 16 + (lane & 15);                   // co
  int k = kg >> 7, ci = kg & 127;
  dst[idx] = (_Float16)w2[(col * 128 + ci) * 5 + k]; // conv2_w[co][ci][k]
}

// ---------------- K1: edge gather + conv1 + ReLU + BN1 + maxpool -> Ph f16 [NS][48][128] ----
__global__ __launch_bounds__(256) void k_conv1(const float* __restrict__ inputs,
                                               const float* __restrict__ w1,
                                               const float* __restrict__ b1,
                                               const float* __restrict__ g,
                                               const float* __restrict__ bb,
                                               const float* __restrict__ mm,
                                               const float* __restrict__ vv,
                                               _Float16* __restrict__ Ph) {
  int n = blockIdx.x;                    // 0..NS-1
  int b = n / NEDGE, e = n % NEDGE;
  int snd, rcv; edge_nodes(e, snd, rcv);
  __shared__ float xs[8][NT];
  const float* sp = inputs + (size_t)(b * NNODE + snd) * NT * NDIM;
  const float* rp = inputs + (size_t)(b * NNODE + rcv) * NT * NDIM;
  for (int i = threadIdx.x; i < 8 * NT; i += 256) {
    int ci = i / NT, t = i % NT;
    xs[ci][t] = (ci < 4) ? sp[t * 4 + ci] : rp[t * 4 + (ci - 4)];
  }
  __syncthreads();
  for (int oi = threadIdx.x; oi < 128 * TP; oi += 256) {
    int co = oi / TP, u = oi % TP;
    float scale = g[co] * rsqrtf(vv[co] + BN_EPS);
    float shift = bb[co] - mm[co] * scale;
    const float* w = w1 + co * 40;
    float best = -1e30f;
#pragma unroll
    for (int tt = 0; tt < 2; ++tt) {
      int t0 = 2 * u + tt;
      float acc = b1[co];
#pragma unroll
      for (int ci = 0; ci < 8; ++ci)
#pragma unroll
        for (int k = 0; k < 5; ++k)
          acc += w[ci * 5 + k] * xs[ci][t0 + k];
      acc = fmaxf(acc, 0.f) * scale + shift;
      best = fmaxf(best, acc);
    }
    Ph[((size_t)n * TP + u) * 128 + co] = (_Float16)best;
  }
}

// ---------------- K2: conv2 as implicit GEMM (WMMA) + ReLU + BN2 -> Z f16 [NS][44][128] -----
__global__ __launch_bounds__(256) void k_conv2(const _Float16* __restrict__ Ph,
                                               const _Float16* __restrict__ W2F,
                                               const float* __restrict__ b2,
                                               const float* __restrict__ g,
                                               const float* __restrict__ bb,
                                               const float* __restrict__ mm,
                                               const float* __restrict__ vv,
                                               _Float16* __restrict__ Z) {
  int mtile = blockIdx.x;
  int lane = threadIdx.x & 31;
  int nt   = threadIdx.x >> 5;           // N-tile 0..7
  int row = mtile * 16 + (lane & 15);
  int n = row / T2, t = row % T2;
  const _Float16* arow = Ph + ((size_t)n * TP + t) * 128;   // time-major, ci contiguous
  const v16h* Bf = (const v16h*)W2F;
  v8f acc = {};
#pragma unroll 4
  for (int c = 0; c < 20; ++c) {         // K = 640 = 20 chunks of 32
    int k = c >> 2, cb = (c & 3) << 5;   // K ordering kg = k*128 + ci
    v16h a  = load_a16(arow + k * 128 + cb, lane);
    v16h bf = Bf[(c * 8 + nt) * 32 + lane];
    acc = wmma16(a, bf, acc);
  }
  int col = nt * 16 + (lane & 15);
  float scale = g[col] * rsqrtf(vv[col] + BN_EPS);
  float shift = bb[col] - mm[col] * scale;
  float bias  = b2[col];
  int rbase = mtile * 16 + ((lane & 16) ? 8 : 0);
#pragma unroll
  for (int i = 0; i < 8; ++i) {
    int rr = rbase + i;
    int nn = rr / T2, tt = rr % T2;
    float val = fmaxf(acc[i] + bias, 0.f) * scale + shift;
    Z[((size_t)nn * T2 + tt) * 128 + col] = (_Float16)val;
  }
}

// ---------------- K3: convp (WMMA) + attn + softmax + weighted temporal mean -> X0h ----------
__global__ __launch_bounds__(256) void k_attnpool(const _Float16* __restrict__ Z,
                                                  const _Float16* __restrict__ WpF,
                                                  const float* __restrict__ bp,
                                                  const float* __restrict__ wa,
                                                  const float* __restrict__ ba,
                                                  _Float16* __restrict__ X0h) {
  int n = blockIdx.x;
  __shared__ float attnw[48];
  __shared__ float logit[T2];
  __shared__ float x0acc[128];
  int tid = threadIdx.x;
  if (tid < 128) x0acc[tid] = 0.f;
  if (tid < T2) {
    const _Float16* zp = Z + ((size_t)n * T2 + tid) * 128;
    float s = ba[0];
    for (int ci = 0; ci < 128; ++ci) s += wa[ci] * (float)zp[ci];
    logit[tid] = s;
  }
  __syncthreads();
  if (tid == 0) {
    float mx = -1e30f;
    for (int t = 0; t < T2; ++t) mx = fmaxf(mx, logit[t]);
    float sum = 0.f;
    for (int t = 0; t < T2; ++t) { float ex = __expf(logit[t] - mx); attnw[t] = ex; sum += ex; }
    float inv = 1.f / (sum * (float)T2);            // softmax * (1/44) mean
    for (int t = 0; t < T2; ++t) attnw[t] *= inv;
    for (int t = T2; t < 48; ++t) attnw[t] = 0.f;
  }
  __syncthreads();
  int lane = tid & 31, nt = tid >> 5;
  int col = nt * 16 + (lane & 15);
  const v16h* Bf = (const v16h*)WpF;
  float bias = bp[col];
  for (int mt = 0; mt < 3; ++mt) {                   // 44 rows padded to 48
    int t  = mt * 16 + (lane & 15);
    int tz = (t < T2) ? t : (T2 - 1);                // clamp (no EXEC divergence); attnw=0 kills pad
    const _Float16* arow = Z + ((size_t)n * T2 + tz) * 128;
    v8f acc = {};
#pragma unroll
    for (int c = 0; c < 4; ++c) {
      v16h a  = load_a16(arow + c * 32, lane);
      v16h bf = Bf[(c * 8 + nt) * 32 + lane];
      acc = wmma16(a, bf, acc);
    }
    int rbase = mt * 16 + ((lane & 16) ? 8 : 0);
    float s = 0.f;
#pragma unroll
    for (int i = 0; i < 8; ++i) s += attnw[rbase + i] * (acc[i] + bias);
    atomicAdd(&x0acc[col], s);
  }
  __syncthreads();
  if (tid < 128) X0h[(size_t)n * 128 + tid] = (_Float16)x0acc[tid];
}

// ---------------- K4: mlp1 (two fused 128x128 WMMA GEMMs + ELU) -> X1f f32, CAT[:,256:384] --
__global__ __launch_bounds__(256) void k_mlp1(const _Float16* __restrict__ X0h,
                                              const _Float16* __restrict__ W1a,
                                              const _Float16* __restrict__ W1b,
                                              const float* __restrict__ b1,
                                              const float* __restrict__ b2,
                                              float* __restrict__ X1f,
                                              _Float16* __restrict__ CAT) {
  __shared__ _Float16 Hh[16 * 128];
  int mtile = blockIdx.x;
  int lane = threadIdx.x & 31, nt = threadIdx.x >> 5;
  int col = nt * 16 + (lane & 15);
  const v16h* Ba = (const v16h*)W1a;
  const v16h* Bb = (const v16h*)W1b;
  {
    const _Float16* arow = X0h + (size_t)(mtile * 16 + (lane & 15)) * 128;
    v8f acc = {};
#pragma unroll
    for (int c = 0; c < 4; ++c) acc = wmma16(load_a16(arow + c * 32, lane), Ba[(c * 8 + nt) * 32 + lane], acc);
    int rbase = (lane & 16) ? 8 : 0;
    float bias = b1[col];
#pragma unroll
    for (int i = 0; i < 8; ++i) Hh[(rbase + i) * 128 + col] = (_Float16)eluf(acc[i] + bias);
  }
  __syncthreads();
  {
    const _Float16* arow = Hh + (lane & 15) * 128;
    v8f acc = {};
#pragma unroll
    for (int c = 0; c < 4; ++c) acc = wmma16(load_a16(arow + c * 32, lane), Bb[(c * 8 + nt) * 32 + lane], acc);
    int rbase = (lane & 16) ? 8 : 0;
    float bias = b2[col];
#pragma unroll
    for (int i = 0; i < 8; ++i) {
      float h = eluf(acc[i] + bias);
      int rr = mtile * 16 + rbase + i;
      X1f[(size_t)rr * 128 + col] = h;                 // for edge2node aggregation
      CAT[(size_t)rr * 384 + 256 + col] = (_Float16)h; // x_skip slot
    }
  }
}

// ---------------- K5: edge2node mean-aggregate + mlp2 (160 rows, VALU) -> X2h ---------------
__global__ __launch_bounds__(128) void k_node(const float* __restrict__ X1f,
                                              const float* __restrict__ w1,
                                              const float* __restrict__ b1,
                                              const float* __restrict__ w2,
                                              const float* __restrict__ b2,
                                              _Float16* __restrict__ X2h) {
  int blk = blockIdx.x;                  // b*40 + node
  int b = blk / NNODE, node = blk % NNODE;
  __shared__ float incv[128], h1v[128];
  int tid = threadIdx.x;
  const float* base = X1f + (size_t)(b * NEDGE + node * (NNODE - 1)) * 128;
  float s = 0.f;
  for (int k = 0; k < NNODE - 1; ++k) s += base[k * 128 + tid];
  incv[tid] = s * (1.f / (float)NNODE);
  __syncthreads();
  float h = b1[tid];
  for (int k = 0; k < 128; ++k) h += incv[k] * w1[k * 128 + tid];
  h1v[tid] = eluf(h);
  __syncthreads();
  float o = b2[tid];
  for (int k = 0; k < 128; ++k) o += h1v[k] * w2[k * 128 + tid];
  X2h[(size_t)blk * 128 + tid] = (_Float16)eluf(o);
}

// ---------------- K6: node2edge gather -> CAT[:,0:256] --------------------------------------
__global__ __launch_bounds__(256) void k_cat(const _Float16* __restrict__ X2h,
                                             _Float16* __restrict__ CAT) {
  int idx = blockIdx.x * 256 + threadIdx.x;
  if (idx >= NS * 128) return;
  int eg = idx >> 7, h = idx & 127;
  int b = eg / NEDGE, e = eg % NEDGE;
  int snd, rcv; edge_nodes(e, snd, rcv);
  CAT[(size_t)eg * 384 + h]       = X2h[(size_t)(b * NNODE + snd) * 128 + h];
  CAT[(size_t)eg * 384 + 128 + h] = X2h[(size_t)(b * NNODE + rcv) * 128 + h];
}

// ---------------- K7: mlp3 (384->128, 128->128 WMMA) + fco (128->2) -> out ------------------
__global__ __launch_bounds__(256) void k_mlp3(const _Float16* __restrict__ CAT,
                                              const _Float16* __restrict__ W3a,
                                              const _Float16* __restrict__ W3b,
                                              const float* __restrict__ b1,
                                              const float* __restrict__ b2,
                                              const float* __restrict__ fw,
                                              const float* __restrict__ fb,
                                              float* __restrict__ out) {
  __shared__ _Float16 Hh[16 * 128];
  __shared__ float H2[16 * 128];
  int mtile = blockIdx.x;
  int lane = threadIdx.x & 31, nt = threadIdx.x >> 5;
  int col = nt * 16 + (lane & 15);
  const v16h* Ba = (const v16h*)W3a;
  const v16h* Bb = (const v16h*)W3b;
  {
    const _Float16* arow = CAT + (size_t)(mtile * 16 + (lane & 15)) * 384;
    v8f acc = {};
#pragma unroll
    for (int c = 0; c < 12; ++c) acc = wmma16(load_a16(arow + c * 32, lane), Ba[(c * 8 + nt) * 32 + lane], acc);
    int rbase = (lane & 16) ? 8 : 0;
    float bias = b1[col];
#pragma unroll
    for (int i = 0; i < 8; ++i) Hh[(rbase + i) * 128 + col] = (_Float16)eluf(acc[i] + bias);
  }
  __syncthreads();
  {
    const _Float16* arow = Hh + (lane & 15) * 128;
    v8f acc = {};
#pragma unroll
    for (int c = 0; c < 4; ++c) acc = wmma16(load_a16(arow + c * 32, lane), Bb[(c * 8 + nt) * 32 + lane], acc);
    int rbase = (lane & 16) ? 8 : 0;
    float bias = b2[col];
#pragma unroll
    for (int i = 0; i < 8; ++i) H2[(rbase + i) * 128 + col] = eluf(acc[i] + bias);
  }
  __syncthreads();
  int tid = threadIdx.x;
  if (tid < 32) {
    int rl = tid >> 1, o = tid & 1;
    float s = fb[o];
    for (int k = 0; k < 128; ++k) s += H2[rl * 128 + k] * fw[k * 2 + o];
    out[(size_t)(mtile * 16 + rl) * 2 + o] = s;
  }
}

extern "C" void kernel_launch(void* const* d_in, const int* in_sizes, int n_in,
                              void* d_out, int out_size, void* d_ws, size_t ws_size,
                              hipStream_t stream) {
  const float* inputs  = (const float*)d_in[0];
  const float* conv1_w = (const float*)d_in[3];
  const float* conv1_b = (const float*)d_in[4];
  const float* bn1_g   = (const float*)d_in[5];
  const float* bn1_b   = (const float*)d_in[6];
  const float* bn1_m   = (const float*)d_in[7];
  const float* bn1_v   = (const float*)d_in[8];
  const float* conv2_w = (const float*)d_in[9];
  const float* conv2_b = (const float*)d_in[10];
  const float* bn2_g   = (const float*)d_in[11];
  const float* bn2_b   = (const float*)d_in[12];
  const float* bn2_m   = (const float*)d_in[13];
  const float* bn2_v   = (const float*)d_in[14];
  const float* convp_w = (const float*)d_in[15];
  const float* convp_b = (const float*)d_in[16];
  const float* conva_w = (const float*)d_in[17];
  const float* conva_b = (const float*)d_in[18];
  const float* mlp1_w1 = (const float*)d_in[19];
  const float* mlp1_b1 = (const float*)d_in[20];
  const float* mlp1_w2 = (const float*)d_in[21];
  const float* mlp1_b2 = (const float*)d_in[22];
  const float* mlp2_w1 = (const float*)d_in[23];
  const float* mlp2_b1 = (const float*)d_in[24];
  const float* mlp2_w2 = (const float*)d_in[25];
  const float* mlp2_b2 = (const float*)d_in[26];
  const float* mlp3_w1 = (const float*)d_in[27];
  const float* mlp3_b1 = (const float*)d_in[28];
  const float* mlp3_w2 = (const float*)d_in[29];
  const float* mlp3_b2 = (const float*)d_in[30];
  const float* fco_w   = (const float*)d_in[31];
  const float* fco_b   = (const float*)d_in[32];
  float* out = (float*)d_out;

  char* ws = (char*)d_ws;
  size_t off = 0;
  auto alloc = [&](size_t bytes) -> void* {
    off = (off + 255) & ~(size_t)255;
    void* p = ws + off;
    off += bytes;
    return p;
  };
  _Float16* Ph  = (_Float16*)alloc((size_t)NS * TP * 128 * 2);
  _Float16* Zb  = (_Float16*)alloc((size_t)NS * T2 * 128 * 2);
  _Float16* X0h = (_Float16*)alloc((size_t)NS * 128 * 2);
  float*    X1f = (float*)   alloc((size_t)NS * 128 * 4);
  _Float16* CAT = (_Float16*)alloc((size_t)NS * 384 * 2);
  _Float16* X2h = (_Float16*)alloc((size_t)NB * NNODE * 128 * 2);
  _Float16* W2F = (_Float16*)alloc((size_t)640 * 128 * 2);
  _Float16* WpF = (_Float16*)alloc((size_t)128 * 128 * 2);
  _Float16* M1a = (_Float16*)alloc((size_t)128 * 128 * 2);
  _Float16* M1b = (_Float16*)alloc((size_t)128 * 128 * 2);
  _Float16* M3a = (_Float16*)alloc((size_t)384 * 128 * 2);
  _Float16* M3b = (_Float16*)alloc((size_t)128 * 128 * 2);

  // weight fragment swizzles
  hipLaunchKernelGGL(k_swz_conv2, dim3(320), dim3(256), 0, stream, conv2_w, W2F);
  hipLaunchKernelGGL(k_swz_tr,    dim3(64),  dim3(256), 0, stream, convp_w, WpF, 128);
  hipLaunchKernelGGL(k_swz_rm,    dim3(64),  dim3(256), 0, stream, mlp1_w1, M1a, 128);
  hipLaunchKernelGGL(k_swz_rm,    dim3(64),  dim3(256), 0, stream, mlp1_w2, M1b, 128);
  hipLaunchKernelGGL(k_swz_rm,    dim3(192), dim3(256), 0, stream, mlp3_w1, M3a, 384);
  hipLaunchKernelGGL(k_swz_rm,    dim3(64),  dim3(256), 0, stream, mlp3_w2, M3b, 128);

  // forward pipeline
  hipLaunchKernelGGL(k_conv1, dim3(NS), dim3(256), 0, stream,
                     inputs, conv1_w, conv1_b, bn1_g, bn1_b, bn1_m, bn1_v, Ph);
  hipLaunchKernelGGL(k_conv2, dim3(MT2), dim3(256), 0, stream,
                     Ph, W2F, conv2_b, bn2_g, bn2_b, bn2_m, bn2_v, Zb);
  hipLaunchKernelGGL(k_attnpool, dim3(NS), dim3(256), 0, stream,
                     Zb, WpF, convp_b, conva_w, conva_b, X0h);
  hipLaunchKernelGGL(k_mlp1, dim3(NS / 16), dim3(256), 0, stream,
                     X0h, M1a, M1b, mlp1_b1, mlp1_b2, X1f, CAT);
  hipLaunchKernelGGL(k_node, dim3(NB * NNODE), dim3(128), 0, stream,
                     X1f, mlp2_w1, mlp2_b1, mlp2_w2, mlp2_b2, X2h);
  hipLaunchKernelGGL(k_cat, dim3((NS * 128) / 256), dim3(256), 0, stream, X2h, CAT);
  hipLaunchKernelGGL(k_mlp3, dim3(NS / 16), dim3(256), 0, stream,
                     CAT, M3a, M3b, mlp3_b1, mlp3_b2, fco_w, fco_b, out);

  (void)in_sizes; (void)n_in; (void)out_size; (void)ws_size;
}
